// SparseAutoEncoder_40192303956344
// MI455X (gfx1250) — matrix operand order repaired and checked by reading, not compiled, not run on hardware
//
#include <hip/hip_runtime.h>
#include <hip/hip_bf16.h>

// ---------------------------------------------------------------------------
// TopK Sparse Autoencoder forward for MI455X (gfx1250, wave32, WMMA).
// Encoder GEMM (549 GFLOP) is compute-bound -> bf16 WMMA (16x16x32) path.
// Tile staging uses GLOBAL_LOAD_ASYNC_TO_LDS_B128 (inline asm; the builtin's
// pointer params use HIP-internal address spaces) + ASYNCcnt double buffering.
// ---------------------------------------------------------------------------

#define B_ROWS 4096
#define D_IN   2048
#define NL     32768   // num latents
#define KSEL   64      // top-k

typedef __bf16 bf16_t;
typedef __attribute__((ext_vector_type(16))) __bf16 v16bf;
typedef __attribute__((ext_vector_type(8)))  __bf16 v8bf;
typedef __attribute__((ext_vector_type(8)))  float  v8f;

#define HAVE_ASYNC_LDS 1

#if defined(__has_builtin)
#if __has_builtin(__builtin_amdgcn_s_wait_asynccnt)
#define HAVE_WAIT_ASYNC 1
#endif
#endif

#if defined(HAVE_ASYNC_LDS)
// Async DMA: global memory -> LDS, 16B per lane, tracked by ASYNCcnt.
// vdst operand is a VGPR holding the LDS byte offset (flat->LDS aperture
// mapping uses addr[31:0]); vaddr is the 64-bit global address (GV mode).
__device__ __forceinline__ void async_copy_b128(const void* g, void* l) {
  unsigned int       loff = (unsigned int)(unsigned long long)l;
  unsigned long long gadr = (unsigned long long)g;
  asm volatile("global_load_async_to_lds_b128 %0, %1, off"
               :
               : "v"(loff), "v"(gadr)
               : "memory");
}
__device__ __forceinline__ void wait_async0() {
#if defined(HAVE_WAIT_ASYNC)
  __builtin_amdgcn_s_wait_asynccnt(0);
#else
  asm volatile("s_wait_asynccnt 0x0" ::: "memory");
#endif
}
#endif

// ---------------- helpers ----------------
__device__ __forceinline__ unsigned short f2bf(float f) {
  unsigned int u = __float_as_uint(f);
  u += 0x7FFFu + ((u >> 16) & 1u);   // round-to-nearest-even
  return (unsigned short)(u >> 16);
}

// ---------------- kernel 1a: x - b_dec -> bf16 ----------------
__global__ __launch_bounds__(256) void conv_x_kernel(const float* __restrict__ x,
                                                     const float* __restrict__ b_dec,
                                                     unsigned short* __restrict__ xb) {
  size_t i4 = (size_t)blockIdx.x * blockDim.x + threadIdx.x;   // one float4 each
  size_t n4 = (size_t)B_ROWS * D_IN / 4;
  if (i4 >= n4) return;
  size_t base = i4 * 4;
  int col = (int)(base % D_IN);
  float4 v = *(const float4*)(x + base);
  float4 b;
  b.x = b_dec[col + 0]; b.y = b_dec[col + 1]; b.z = b_dec[col + 2]; b.w = b_dec[col + 3];
  unsigned long long p =  (unsigned long long)f2bf(v.x - b.x)
                       | ((unsigned long long)f2bf(v.y - b.y) << 16)
                       | ((unsigned long long)f2bf(v.z - b.z) << 32)
                       | ((unsigned long long)f2bf(v.w - b.w) << 48);
  *(unsigned long long*)(xb + base) = p;
}

// ---------------- kernel 1b: W_enc -> bf16 ----------------
__global__ __launch_bounds__(256) void conv_w_kernel(const float* __restrict__ w,
                                                     unsigned short* __restrict__ wb) {
  size_t i4 = (size_t)blockIdx.x * blockDim.x + threadIdx.x;
  size_t n4 = (size_t)NL * D_IN / 4;
  if (i4 >= n4) return;
  size_t base = i4 * 4;
  float4 v = *(const float4*)(w + base);
  unsigned long long p =  (unsigned long long)f2bf(v.x)
                       | ((unsigned long long)f2bf(v.y) << 16)
                       | ((unsigned long long)f2bf(v.z) << 32)
                       | ((unsigned long long)f2bf(v.w) << 48);
  *(unsigned long long*)(wb + base) = p;
}

// ---------------- kernel 2: column sums + sum of squares (for variance) ----
__global__ __launch_bounds__(256) void col_stats_kernel(const float* __restrict__ x,
                                                        float* __restrict__ colsum,
                                                        float* __restrict__ sumsq) {
  int d = blockIdx.x * 256 + threadIdx.x;   // 8 blocks cover 2048 cols
  if (d >= D_IN) return;
  float s = 0.f, s2 = 0.f;
  for (int b = 0; b < B_ROWS; ++b) {
    float v = x[(size_t)b * D_IN + d];
    s += v; s2 += v * v;
  }
  colsum[d] = s;
  atomicAdd(sumsq, s2);
}

// ---------------- kernel 3: encoder GEMM (bf16 WMMA) + bias + relu ---------
// C[m,n] = relu( sum_k xb[m,k] * wb[n,k] + b_enc[n] ), tiles 128x128, K-step 32.
#define TM 128
#define TN 128
#define TK 32
#define LSTR 40   // padded LDS row stride (ushorts) to spread banks

__global__ __launch_bounds__(256) void enc_gemm_kernel(const unsigned short* __restrict__ xb,
                                                       const unsigned short* __restrict__ wb,
                                                       const float* __restrict__ b_enc,
                                                       float* __restrict__ latents) {
  const int tid  = threadIdx.x;
  const int lane = tid & 31;
  const int wave = tid >> 5;                 // 0..7
  const int m0 = blockIdx.y * TM;
  const int n0 = blockIdx.x * TN;
  const int wm = (wave & 1) * 64;            // wave M offset (2 waves in M)
  const int wn = (wave >> 1) * 32;           // wave N offset (4 waves in N)
  const int lm = lane & 15;
  const int kg = lane >> 4;                  // K-group (ISA 16-bit fragment layout)

  v8f acc[4][2];
#pragma unroll
  for (int i = 0; i < 4; ++i)
#pragma unroll
    for (int j = 0; j < 2; ++j)
      acc[i][j] = (v8f){0.f, 0.f, 0.f, 0.f, 0.f, 0.f, 0.f, 0.f};

#if defined(HAVE_ASYNC_LDS)
  // ------- double-buffered async (ASYNCcnt) staging -------
  __shared__ unsigned short sa[2][TM][LSTR];
  __shared__ unsigned short sb[2][TN][LSTR];

  const int srow = lane >> 2;        // 0..7 : row within 8-row slab
  const int schk = (lane & 3) * 8;   // 16B chunk offset in ushorts

  auto stage = [&](int kt, int bufi) {
#pragma unroll
    for (int t = 0; t < 2; ++t) {
      int r = wave * 16 + t * 8 + srow;    // this wave covers rows [wave*16, +16)
      async_copy_b128(xb + (size_t)(m0 + r) * D_IN + kt + schk, &sa[bufi][r][schk]);
      async_copy_b128(wb + (size_t)(n0 + r) * D_IN + kt + schk, &sb[bufi][r][schk]);
    }
  };

  stage(0, 0);
  wait_async0();
  __syncthreads();

  int buf = 0;
  for (int kt = 0; kt < D_IN; kt += TK) {
    if (kt + TK < D_IN) stage(kt + TK, buf ^ 1);   // prefetch next K-tile

    // B fragments: lane holds col n = wn + j*16 + lm, K = {kg*8..+7, kg*8+16..+23}
    union { v16bf v; v8bf h[2]; } bf[2];
#pragma unroll
    for (int j = 0; j < 2; ++j) {
      const unsigned short* p = &sb[buf][wn + j * 16 + lm][kg * 8];
      bf[j].h[0] = *(const v8bf*)(p);
      bf[j].h[1] = *(const v8bf*)(p + 16);
    }
#pragma unroll
    for (int i = 0; i < 4; ++i) {
      // A fragment: lane holds row m = wm + i*16 + lm, K = {kg*8..+7, kg*8+16..+23}
      union { v16bf v; v8bf h[2]; } af;
      const unsigned short* pa = &sa[buf][wm + i * 16 + lm][0];
      af.h[0] = *(const v8bf*)(pa + kg * 8);
      af.h[1] = *(const v8bf*)(pa + kg * 8 + 16);
#pragma unroll
      for (int j = 0; j < 2; ++j) {
        acc[i][j] = __builtin_amdgcn_wmma_f32_16x16x32_bf16(
            false, af.v, false, bf[j].v, (short)0, acc[i][j], false, false);
      }
    }
    wait_async0();       // own async loads into next buffer done
    __syncthreads();     // everyone's loads visible; everyone done reading cur
    buf ^= 1;
  }
#else
  // ------- synchronous fallback staging -------
  __shared__ unsigned short sa1[TM][LSTR];
  __shared__ unsigned short sb1[TN][LSTR];
  const int lrow = tid >> 3;         // 0..31
  const int lcol = (tid & 7) * 4;    // bf16 col within 32, step 4 (8 bytes)

  for (int kt = 0; kt < D_IN; kt += TK) {
#pragma unroll
    for (int rr = 0; rr < TM; rr += 32) {
      int r = rr + lrow;
      unsigned long long va = *(const unsigned long long*)(xb + (size_t)(m0 + r) * D_IN + kt + lcol);
      *(unsigned long long*)(&sa1[r][lcol]) = va;
      unsigned long long vb = *(const unsigned long long*)(wb + (size_t)(n0 + r) * D_IN + kt + lcol);
      *(unsigned long long*)(&sb1[r][lcol]) = vb;
    }
    __syncthreads();

    union { v16bf v; v8bf h[2]; } bf[2];
#pragma unroll
    for (int j = 0; j < 2; ++j) {
      const unsigned short* p = &sb1[wn + j * 16 + lm][kg * 16];
      bf[j].h[0] = *(const v8bf*)(p);
      bf[j].h[1] = *(const v8bf*)(p + 8);
    }
#pragma unroll
    for (int i = 0; i < 4; ++i) {
      union { v16bf v; v8bf h[2]; } af;
      const unsigned short* pa = &sa1[wm + i * 16 + lm][0];
      af.h[0] = *(const v8bf*)(pa + kg * 8);
      af.h[1] = *(const v8bf*)(pa + kg * 8 + 16);
#pragma unroll
      for (int j = 0; j < 2; ++j) {
        acc[i][j] = __builtin_amdgcn_wmma_f32_16x16x32_bf16(
            false, af.v, false, bf[j].v, (short)0, acc[i][j], false, false);
      }
    }
    __syncthreads();
  }
#endif

  // epilogue: C layout — VGPR r: M = base + (lane>>4)*8 + r, N = base + (lane&15)
#pragma unroll
  for (int i = 0; i < 4; ++i) {
#pragma unroll
    for (int j = 0; j < 2; ++j) {
      int colb = n0 + wn + j * 16 + (lane & 15);
      float bias = b_enc[colb];
#pragma unroll
      for (int r = 0; r < 8; ++r) {
        int row = m0 + wm + i * 16 + (lane >> 4) * 8 + r;
        float v = acc[i][j][r] + bias;
        v = v > 0.f ? v : 0.f;
        latents[(size_t)row * NL + colb] = v;
      }
    }
  }
}

// ---------------- kernel 4: per-row radix-select top-64 + bitonic sort -----
#define CAP 3072
__global__ __launch_bounds__(256) void topk_kernel(const float* __restrict__ latents,
                                                   float* __restrict__ top_acts,
                                                   int* __restrict__ top_idx) {
  const int row = blockIdx.x;
  const int tid = threadIdx.x;
  const float* __restrict__ v = latents + (size_t)row * NL;

  __shared__ unsigned int hist[256];
  __shared__ float cval0[CAP]; __shared__ int cidx0[CAP];
  __shared__ float cval1[CAP]; __shared__ int cidx1[CAP];
  __shared__ float oval[KSEL]; __shared__ int oidx[KSEL];
  __shared__ unsigned int out_cnt, cand_cnt, next_cnt, s_bucket, s_remaining;

  hist[tid] = 0u;
  if (tid == 0) { out_cnt = 0u; cand_cnt = 0u; }
  __syncthreads();

  // level 0: MSB histogram over full row (values >= 0 -> bits monotonic)
  for (int i = tid; i < NL; i += 256)
    atomicAdd(&hist[__float_as_uint(v[i]) >> 24], 1u);
  __syncthreads();
  if (tid == 0) {
    unsigned int cum = 0; int b = 255;
    for (; b > 0; --b) { unsigned int c = hist[b]; if (cum + c >= (unsigned)KSEL) break; cum += c; }
    s_bucket = (unsigned)b;
    s_remaining = (unsigned)KSEL - cum;
  }
  __syncthreads();
  unsigned int bsel = s_bucket;
  unsigned int remaining = s_remaining;

  // emit sure winners, compact boundary-bucket candidates to LDS
  for (int i = tid; i < NL; i += 256) {
    float f = v[i];
    unsigned int tb = __float_as_uint(f) >> 24;
    if (tb > bsel) {
      unsigned int p = atomicAdd(&out_cnt, 1u);
      if (p < (unsigned)KSEL) { oval[p] = f; oidx[p] = i; }
    } else if (tb == bsel) {
      unsigned int p = atomicAdd(&cand_cnt, 1u);
      if (p < (unsigned)CAP) { cval0[p] = f; cidx0[p] = i; }
    }
  }
  __syncthreads();
  unsigned int ncand = cand_cnt < (unsigned)CAP ? cand_cnt : (unsigned)CAP;

  float* curv = cval0; int* curi = cidx0;
  float* nxtv = cval1; int* nxti = cidx1;

  for (int shift = 16; shift >= 0 && remaining < ncand; shift -= 8) {
    hist[tid] = 0u;
    if (tid == 0) next_cnt = 0u;
    __syncthreads();
    for (unsigned int i = tid; i < ncand; i += 256)
      atomicAdd(&hist[(__float_as_uint(curv[i]) >> shift) & 255u], 1u);
    __syncthreads();
    if (tid == 0) {
      unsigned int cum = 0; int b = 255;
      for (; b > 0; --b) { unsigned int c = hist[b]; if (cum + c >= remaining) break; cum += c; }
      s_bucket = (unsigned)b;
      s_remaining = remaining - cum;
    }
    __syncthreads();
    bsel = s_bucket;
    for (unsigned int i = tid; i < ncand; i += 256) {
      float f = curv[i];
      unsigned int by = (__float_as_uint(f) >> shift) & 255u;
      if (by > bsel) {
        unsigned int p = atomicAdd(&out_cnt, 1u);
        if (p < (unsigned)KSEL) { oval[p] = f; oidx[p] = curi[i]; }
      } else if (by == bsel) {
        unsigned int p = atomicAdd(&next_cnt, 1u);
        nxtv[p] = f; nxti[p] = curi[i];
      }
    }
    __syncthreads();
    ncand = next_cnt;
    remaining = s_remaining;
    float* tv = curv; curv = nxtv; nxtv = tv;
    int*   ti = curi; curi = nxti; nxti = ti;
    __syncthreads();
  }

  // survivors are ties (or everything selected): emit `remaining` of them
  for (unsigned int i = tid; i < remaining; i += 256) {
    unsigned int j = (ncand > 0u) ? (i % ncand) : 0u;
    unsigned int p = atomicAdd(&out_cnt, 1u);
    if (p < (unsigned)KSEL) {
      oval[p] = (ncand > 0u) ? curv[j] : 0.f;
      oidx[p] = (ncand > 0u) ? curi[j] : 0;
    }
  }
  __syncthreads();
  for (unsigned int i = out_cnt + tid; i < (unsigned)KSEL; i += 256) { oval[i] = 0.f; oidx[i] = 0; }
  __syncthreads();

  // bitonic sort 64 elems ascending, write reversed (jax top_k is sorted desc)
  for (int ksz = 2; ksz <= KSEL; ksz <<= 1) {
    for (int j = ksz >> 1; j > 0; j >>= 1) {
      __syncthreads();
      if (tid < KSEL) {
        int ixj = tid ^ j;
        if (ixj > tid) {
          bool up = ((tid & ksz) == 0);
          float a = oval[tid], b2 = oval[ixj];
          if ((a > b2) == up) {
            oval[tid] = b2; oval[ixj] = a;
            int t = oidx[tid]; oidx[tid] = oidx[ixj]; oidx[ixj] = t;
          }
        }
      }
    }
  }
  __syncthreads();
  if (tid < KSEL) {
    top_acts[(size_t)row * KSEL + tid] = oval[KSEL - 1 - tid];
    top_idx [(size_t)row * KSEL + tid] = oidx[KSEL - 1 - tid];
  }
}

// ---------------- kernel 5: sparse decode + fused L2 error -----------------
__global__ __launch_bounds__(256) void decode_kernel(const float* __restrict__ top_acts,
                                                     const int* __restrict__ top_idx,
                                                     const float* __restrict__ W_dec,
                                                     const float* __restrict__ b_dec,
                                                     const float* __restrict__ x,
                                                     float* __restrict__ sae_out,
                                                     float* __restrict__ l2) {
  const int row = blockIdx.x;
  const int tid = threadIdx.x;
  __shared__ float sa[KSEL];
  __shared__ int   si[KSEL];
  __shared__ float red[256];
  if (tid < KSEL) {
    sa[tid] = top_acts[(size_t)row * KSEL + tid];
    si[tid] = top_idx [(size_t)row * KSEL + tid];
  }
  __syncthreads();

  float acc[8];
#pragma unroll
  for (int c = 0; c < 8; ++c) acc[c] = 0.f;
  for (int j = 0; j < KSEL; ++j) {
    if (j + 1 < KSEL)   // prefetch next gathered row (global_prefetch_b8)
      __builtin_prefetch(W_dec + (size_t)si[j + 1] * D_IN + tid, 0, 1);
    float a = sa[j];
    const float* wrow = W_dec + (size_t)si[j] * D_IN;
#pragma unroll
    for (int c = 0; c < 8; ++c) acc[c] += a * wrow[tid + c * 256];
  }
  float e2 = 0.f;
#pragma unroll
  for (int c = 0; c < 8; ++c) {
    int col = tid + c * 256;
    float o = acc[c] + b_dec[col];
    sae_out[(size_t)row * D_IN + col] = o;
    float e = o - x[(size_t)row * D_IN + col];
    e2 += e * e;
  }
  red[tid] = e2;
  __syncthreads();
  for (int s = 128; s > 0; s >>= 1) {
    if (tid < s) red[tid] += red[tid + s];
    __syncthreads();
  }
  if (tid == 0) atomicAdd(l2, red[0]);
}

// ---------------- kernel 6: loss = l2 / total_variance ---------------------
__global__ __launch_bounds__(256) void finalize_kernel(const float* __restrict__ colsum,
                                                       const float* __restrict__ sumsq,
                                                       const float* __restrict__ l2,
                                                       float* __restrict__ loss) {
  __shared__ float red[256];
  int tid = threadIdx.x;
  float s = 0.f;
  for (int d = tid; d < D_IN; d += 256) { float c = colsum[d]; s += c * c; }
  red[tid] = s;
  __syncthreads();
  for (int st = 128; st > 0; st >>= 1) {
    if (tid < st) red[tid] += red[tid + st];
    __syncthreads();
  }
  if (tid == 0) {
    float tv = *sumsq - red[0] / (float)B_ROWS;
    *loss = *l2 / tv;
  }
}

// ---------------- launcher -------------------------------------------------
extern "C" void kernel_launch(void* const* d_in, const int* in_sizes, int n_in,
                              void* d_out, int out_size, void* d_ws, size_t ws_size,
                              hipStream_t stream) {
  const float* x     = (const float*)d_in[0];
  const float* W_enc = (const float*)d_in[1];
  const float* b_enc = (const float*)d_in[2];
  const float* W_dec = (const float*)d_in[3];
  const float* b_dec = (const float*)d_in[4];
  // d_in[5] = k (compile-time 64)

  float* out      = (float*)d_out;
  float* sae_out  = out;                                         // [B, D_IN]
  float* top_acts = out + (size_t)B_ROWS * D_IN;                 // [B, K]
  int*   top_idx  = (int*)(out + (size_t)B_ROWS * D_IN + (size_t)B_ROWS * KSEL);
  float* loss     = out + (size_t)B_ROWS * D_IN + 2 * (size_t)B_ROWS * KSEL;

  char* ws = (char*)d_ws;
  const size_t XB_BYTES  = (size_t)B_ROWS * D_IN * 2;   // 16 MB
  const size_t WB_BYTES  = (size_t)NL * D_IN * 2;       // 128 MB
  const size_t LAT_BYTES = (size_t)B_ROWS * NL * 4;     // 512 MB
  unsigned short* xb      = (unsigned short*)ws;
  unsigned short* wb      = (unsigned short*)(ws + XB_BYTES);
  float*          latents = (float*)(ws + XB_BYTES + WB_BYTES);
  float*          colsum  = (float*)(ws + XB_BYTES + WB_BYTES + LAT_BYTES);
  float*          sumsq   = colsum + D_IN;
  float*          l2      = sumsq + 1;

  (void)hipMemsetAsync(sumsq, 0, 2 * sizeof(float), stream);   // sumsq + l2

  {
    size_t n4 = (size_t)B_ROWS * D_IN / 4;
    conv_x_kernel<<<(unsigned)((n4 + 255) / 256), 256, 0, stream>>>(x, b_dec, xb);
  }
  {
    size_t n4 = (size_t)NL * D_IN / 4;
    conv_w_kernel<<<(unsigned)((n4 + 255) / 256), 256, 0, stream>>>(W_enc, wb);
  }
  col_stats_kernel<<<D_IN / 256, 256, 0, stream>>>(x, colsum, sumsq);

  enc_gemm_kernel<<<dim3(NL / TN, B_ROWS / TM), 256, 0, stream>>>(xb, wb, b_enc, latents);

  topk_kernel<<<B_ROWS, 256, 0, stream>>>(latents, top_acts, top_idx);

  decode_kernel<<<B_ROWS, 256, 0, stream>>>(top_acts, top_idx, W_dec, b_dec, x, sae_out, l2);

  finalize_kernel<<<1, 256, 0, stream>>>(colsum, sumsq, l2, loss);
}
